// EventDrivenFusionFast_45217415692848
// MI455X (gfx1250) — matrix operand order repaired and checked by reading, not compiled, not run on hardware
//
#include <hip/hip_runtime.h>

// ---------------------------------------------------------------------------
// EventDrivenFusion for MI455X (gfx1250, wave32, WMMA 16x16x32 f16->f32)
// B=8, T=2048, D=128, NH=4, HD=32.  All matmuls via v_wmma_f32_16x16x32_f16.
// ---------------------------------------------------------------------------

#define BB  8
#define TT  2048
#define DD  128
#define NHH 4
#define HDD 32

typedef _Float16 v16h __attribute__((ext_vector_type(16)));
typedef _Float16 h8   __attribute__((ext_vector_type(8)));
typedef float    v8f  __attribute__((ext_vector_type(8)));

// Load a 16-half WMMA fragment as two contiguous 16-byte LDS reads.
static __device__ __forceinline__ v16h load_frag16(const _Float16* p0,
                                                   const _Float16* p1) {
  v16h r;
  *reinterpret_cast<h8*>(&r)       = *reinterpret_cast<const h8*>(p0);
  *(reinterpret_cast<h8*>(&r) + 1) = *reinterpret_cast<const h8*>(p1);
  return r;
}

enum { MODE_NONE = 0, MODE_LN_GELU = 1, MODE_LN_TANH_MASK = 2, MODE_RESID_LN = 3 };

// ---------------------------------------------------------------------------
// Fused GEMM + epilogue.
//   out[M,128] = epilogue( concat(A0[:,0:K0], A1[:,0:K1]) @ W^T + bias )
//   W is [128, K0+K1] row-major (PyTorch-linear style).
// Block: 256 threads (8 waves). Each block does 128 rows x 128 cols.
// Each wave owns a 16-row strip and 8 N-tiles; K consumed in chunks of 32.
// LDS layouts (stride 40 halves = 80B, keeps 16B alignment of fragments):
//   sA[row][k]   (activations, row-major)
//   sB[n][k]     (weights, already "transposed" so B fragments are contiguous)
// Epilogue fully in registers; per-row LayerNorm via shfl_xor within the
// 16-lane group that holds those rows in the WMMA C/D layout.
// ---------------------------------------------------------------------------
__global__ __launch_bounds__(256)
void gemm_ln_kernel(const float* __restrict__ A0, const float* __restrict__ A1,
                    int K0, int K1,
                    const float* __restrict__ W, const float* __restrict__ bias,
                    const float* __restrict__ gamma, const float* __restrict__ beta,
                    const float* __restrict__ resid, const float* __restrict__ mask,
                    float* __restrict__ out, int mode)
{
  __shared__ __align__(16) _Float16 sA[128 * 40];
  __shared__ __align__(16) _Float16 sB[128 * 40];

  const int  tid  = threadIdx.x;
  const int  wave = tid >> 5;
  const int  lane = tid & 31;
  const int  l    = lane & 15;
  const bool hi   = lane >= 16;
  const int  K    = K0 + K1;
  const long rowBase = (long)blockIdx.x * 128;

  v8f acc[8] = {};

  for (int kc = 0; kc < K; kc += 32) {
    __syncthreads();
    {
      const int  r    = tid >> 1;          // 0..127  (row for A, n for W)
      const int  kk   = (tid & 1) << 4;    // 0 or 16
      const long grow = rowBase + r;
      const int  kg   = kc + kk;
      const float* asrc = (kg < K0) ? (A0 + grow * K0 + kg)
                                    : (A1 + grow * K1 + (kg - K0));
      const float* wsrc = W + (long)r * K + kg;
#pragma unroll
      for (int j = 0; j < 16; j += 4) {
        float4 fa = *reinterpret_cast<const float4*>(asrc + j);
        float4 fw = *reinterpret_cast<const float4*>(wsrc + j);
        sA[r * 40 + kk + j + 0] = (_Float16)fa.x;
        sA[r * 40 + kk + j + 1] = (_Float16)fa.y;
        sA[r * 40 + kk + j + 2] = (_Float16)fa.z;
        sA[r * 40 + kk + j + 3] = (_Float16)fa.w;
        sB[r * 40 + kk + j + 0] = (_Float16)fw.x;
        sB[r * 40 + kk + j + 1] = (_Float16)fw.y;
        sB[r * 40 + kk + j + 2] = (_Float16)fw.z;
        sB[r * 40 + kk + j + 3] = (_Float16)fw.w;
      }
    }
    __syncthreads();

    // A fragment: lane<16 -> K 0..7 & 16..23 ; lane>=16 -> K 8..15 & 24..31
    const int arow = wave * 16 + l;
    const int c0   = hi ? 8 : 0;
    v16h af = load_frag16(&sA[arow * 40 + c0], &sA[arow * 40 + c0 + 16]);

    // B fragment: lane<16 -> K 0..7 & 16..23 ; lane>=16 -> K 8..15 & 24..31 (col = n)
    const int kb0 = hi ? 8 : 0;
#pragma unroll
    for (int nt = 0; nt < 8; nt++) {
      const int bn = nt * 16 + l;
      v16h bf = load_frag16(&sB[bn * 40 + kb0], &sB[bn * 40 + kb0 + 16]);
      acc[nt] = __builtin_amdgcn_wmma_f32_16x16x32_f16(
          false, af, false, bf, (short)0, acc[nt], false, false);
    }
  }

  // ---- epilogue (register resident) ----
  float bcol[8], gcol[8], btc[8];
#pragma unroll
  for (int nt = 0; nt < 8; nt++) {
    bcol[nt] = bias[nt * 16 + l];
    gcol[nt] = (mode != MODE_NONE) ? gamma[nt * 16 + l] : 1.f;
    btc[nt]  = (mode != MODE_NONE) ? beta[nt * 16 + l]  : 0.f;
  }

  const int rlocal = wave * 16 + (hi ? 8 : 0);
#pragma unroll
  for (int r = 0; r < 8; r++) {
    const long grow = rowBase + rlocal + r;
#pragma unroll
    for (int nt = 0; nt < 8; nt++) {
      float x = acc[nt][r] + bcol[nt];
      if (mode == MODE_RESID_LN) x += resid[grow * 128 + nt * 16 + l];
      acc[nt][r] = x;
    }
    if (mode != MODE_NONE) {
      float s = 0.f, ss = 0.f;
#pragma unroll
      for (int nt = 0; nt < 8; nt++) { float x = acc[nt][r]; s += x; ss += x * x; }
      s  += __shfl_xor(s, 1);  s  += __shfl_xor(s, 2);
      s  += __shfl_xor(s, 4);  s  += __shfl_xor(s, 8);
      ss += __shfl_xor(ss, 1); ss += __shfl_xor(ss, 2);
      ss += __shfl_xor(ss, 4); ss += __shfl_xor(ss, 8);
      const float mean = s * (1.f / 128.f);
      const float var  = ss * (1.f / 128.f) - mean * mean;
      const float rstd = rsqrtf(var + 1e-5f);
      const float mk   = (mode == MODE_LN_TANH_MASK) ? mask[grow] : 1.f;
#pragma unroll
      for (int nt = 0; nt < 8; nt++) {
        float y = (acc[nt][r] - mean) * rstd * gcol[nt] + btc[nt];
        if (mode == MODE_LN_GELU)           y = 0.5f * y * (1.f + erff(y * 0.70710678118f));
        else if (mode == MODE_LN_TANH_MASK) y = tanhf(y) * mk;
        acc[nt][r] = y;
      }
    }
#pragma unroll
    for (int nt = 0; nt < 8; nt++)
      out[grow * 128 + nt * 16 + l] = acc[nt][r];
  }
}

// ---------------------------------------------------------------------------
// Per-channel EMA scan along T (in place). 1024 threads = B*D sequences,
// consecutive threads -> consecutive channels -> coalesced per timestep.
// ---------------------------------------------------------------------------
__global__ __launch_bounds__(256)
void scan_kernel(float* __restrict__ buf, const float* __restrict__ decay)
{
  const int tid = blockIdx.x * 256 + threadIdx.x;   // 0..1023
  const int b   = tid >> 7;
  const int ch  = tid & 127;
  const float d = 1.f / (1.f + __expf(-decay[ch]));
  float s = 0.f;
  long idx = (long)b * TT * DD + ch;
  for (int t = 0; t < TT; t++) { s = d * s + buf[idx]; buf[idx] = s; idx += DD; }
}

// ---------------------------------------------------------------------------
// Flash attention. Grid (B*NH, T/128); block = 256 threads = 8 waves.
// Each wave: 16 query rows x HD=32 output.  Key chunks of 32 -> per chunk:
//   2 WMMAs for S = Q K^T (K-dim = HD = 32, exactly one 16x16x32 op per tile)
//   online softmax (max/sum via shfl_xor within 16-lane group)
//   2 WMMAs for O += P V   (P transposed via per-wave private LDS tile)
// ---------------------------------------------------------------------------
__global__ __launch_bounds__(256)
void attn_kernel(const float* __restrict__ qb, const float* __restrict__ kb,
                 const float* __restrict__ vb, float* __restrict__ ctx)
{
  __shared__ __align__(16) _Float16 sQ [128 * 40];  // [qrow][hd], pre-scaled
  __shared__ __align__(16) _Float16 sK [ 32 * 40];  // [key][hd]
  __shared__ __align__(16) _Float16 sVt[ 32 * 40];  // [hd][key] (transposed)
  __shared__ __align__(16) _Float16 sP [128 * 40];  // per-wave P staging

  const int  tid  = threadIdx.x;
  const int  wave = tid >> 5;
  const int  lane = tid & 31;
  const int  l    = lane & 15;
  const bool hi   = lane >= 16;
  const int  b    = blockIdx.x >> 2;
  const int  h    = blockIdx.x & 3;
  const int  qbase = blockIdx.y * 128;
  const long hoff  = (long)h * HDD;

  // Stage Q tile (scaled by 1/sqrt(HD))
  {
    const int r  = tid >> 1;
    const int kk = (tid & 1) << 4;
    const float* src = qb + ((long)(b * TT) + qbase + r) * DD + hoff + kk;
#pragma unroll
    for (int j = 0; j < 16; j += 4) {
      float4 f = *reinterpret_cast<const float4*>(src + j);
      sQ[r * 40 + kk + j + 0] = (_Float16)(f.x * 0.17677669529f);
      sQ[r * 40 + kk + j + 1] = (_Float16)(f.y * 0.17677669529f);
      sQ[r * 40 + kk + j + 2] = (_Float16)(f.z * 0.17677669529f);
      sQ[r * 40 + kk + j + 3] = (_Float16)(f.w * 0.17677669529f);
    }
  }
  __syncthreads();

  const int arow = wave * 16 + l;
  const int c0   = hi ? 8 : 0;
  const int kb0  = hi ? 8 : 0;
  const v16h aq  = load_frag16(&sQ[arow * 40 + c0], &sQ[arow * 40 + c0 + 16]);

  float m_run[8], l_run[8];
#pragma unroll
  for (int r = 0; r < 8; r++) { m_run[r] = -1e30f; l_run[r] = 0.f; }
  v8f o0 = {}, o1 = {};

  for (int kc = 0; kc < TT; kc += 32) {
    __syncthreads();
    {
      const int r  = tid >> 3;          // 0..31 key row
      const int kk = (tid & 7) << 2;    // 0..28
      const long base = ((long)(b * TT) + kc + r) * DD + hoff + kk;
      float4 kf = *reinterpret_cast<const float4*>(kb + base);
      float4 vf = *reinterpret_cast<const float4*>(vb + base);
      sK [r * 40 + kk + 0] = (_Float16)kf.x;
      sK [r * 40 + kk + 1] = (_Float16)kf.y;
      sK [r * 40 + kk + 2] = (_Float16)kf.z;
      sK [r * 40 + kk + 3] = (_Float16)kf.w;
      sVt[(kk + 0) * 40 + r] = (_Float16)vf.x;
      sVt[(kk + 1) * 40 + r] = (_Float16)vf.y;
      sVt[(kk + 2) * 40 + r] = (_Float16)vf.z;
      sVt[(kk + 3) * 40 + r] = (_Float16)vf.w;
    }
    __syncthreads();

    // S = Q K^T  (B fragment col = key, K-dim = hd)
    v8f s0 = {}, s1 = {};
    {
      v16h bk0 = load_frag16(&sK[l * 40 + kb0], &sK[l * 40 + kb0 + 16]);
      s0 = __builtin_amdgcn_wmma_f32_16x16x32_f16(false, aq, false, bk0,
                                                  (short)0, s0, false, false);
      v16h bk1 = load_frag16(&sK[(16 + l) * 40 + kb0], &sK[(16 + l) * 40 + kb0 + 16]);
      s1 = __builtin_amdgcn_wmma_f32_16x16x32_f16(false, aq, false, bk1,
                                                  (short)0, s1, false, false);
    }

    // Online softmax update; write P (f16) into this wave's LDS tile.
    const int prow0 = wave * 16 + (hi ? 8 : 0);
#pragma unroll
    for (int r = 0; r < 8; r++) {
      float tm = fmaxf(s0[r], s1[r]);
      tm = fmaxf(tm, __shfl_xor(tm, 1));
      tm = fmaxf(tm, __shfl_xor(tm, 2));
      tm = fmaxf(tm, __shfl_xor(tm, 4));
      tm = fmaxf(tm, __shfl_xor(tm, 8));
      const float mn = fmaxf(m_run[r], tm);
      const float sc = __expf(m_run[r] - mn);
      m_run[r] = mn;
      const float e0 = __expf(s0[r] - mn);
      const float e1 = __expf(s1[r] - mn);
      float ps = e0 + e1;
      ps += __shfl_xor(ps, 1); ps += __shfl_xor(ps, 2);
      ps += __shfl_xor(ps, 4); ps += __shfl_xor(ps, 8);
      l_run[r] = l_run[r] * sc + ps;
      o0[r] *= sc; o1[r] *= sc;
      sP[(prow0 + r) * 40 + l]      = (_Float16)e0;
      sP[(prow0 + r) * 40 + 16 + l] = (_Float16)e1;
    }
    // LDS is in-order within a wave; just stop the compiler reordering.
    asm volatile("" ::: "memory");

    // O += P V   (A = P 16x32 from LDS, B fragment col = hd, K-dim = key)
    const v16h ap = load_frag16(&sP[arow * 40 + c0], &sP[arow * 40 + c0 + 16]);
    {
      v16h bv0 = load_frag16(&sVt[l * 40 + kb0], &sVt[l * 40 + kb0 + 16]);
      o0 = __builtin_amdgcn_wmma_f32_16x16x32_f16(false, ap, false, bv0,
                                                  (short)0, o0, false, false);
      v16h bv1 = load_frag16(&sVt[(16 + l) * 40 + kb0], &sVt[(16 + l) * 40 + kb0 + 16]);
      o1 = __builtin_amdgcn_wmma_f32_16x16x32_f16(false, ap, false, bv1,
                                                  (short)0, o1, false, false);
    }
  }

  // Finalize and write ctx (interleaved back into [B,T,D])
  const int r0 = wave * 16 + (hi ? 8 : 0);
#pragma unroll
  for (int r = 0; r < 8; r++) {
    const float inv = 1.f / l_run[r];
    const long  row = (long)(b * TT) + qbase + r0 + r;
    ctx[row * DD + hoff + l]      = o0[r] * inv;
    ctx[row * DD + hoff + 16 + l] = o1[r] * inv;
  }
}

// ---------------------------------------------------------------------------
// Host-side orchestration
// ---------------------------------------------------------------------------
extern "C" void kernel_launch(void* const* d_in, const int* in_sizes, int n_in,
                              void* d_out, int out_size, void* d_ws, size_t ws_size,
                              hipStream_t stream)
{
  const float* lob_feat   = (const float*)d_in[0];
  const float* trade_feat = (const float*)d_in[1];
  const float* has_trade  = (const float*)d_in[2];
  const float* ei_w       = (const float*)d_in[3];
  const float* ei_b       = (const float*)d_in[4];
  const float* ei_g       = (const float*)d_in[5];
  const float* ei_be      = (const float*)d_in[6];
  const float* decay      = (const float*)d_in[7];
  const float* in_proj_w  = (const float*)d_in[8];
  const float* in_proj_b  = (const float*)d_in[9];
  const float* out_proj_w = (const float*)d_in[10];
  const float* out_proj_b = (const float*)d_in[11];
  const float* cn_g       = (const float*)d_in[12];
  const float* cn_b       = (const float*)d_in[13];
  const float* lob_w      = (const float*)d_in[14];
  const float* lob_b      = (const float*)d_in[15];
  const float* lob_g      = (const float*)d_in[16];
  const float* lob_be     = (const float*)d_in[17];
  const float* tr_w       = (const float*)d_in[18];
  const float* tr_b       = (const float*)d_in[19];
  const float* tr_g       = (const float*)d_in[20];
  const float* tr_be      = (const float*)d_in[21];
  const float* of_w       = (const float*)d_in[22];
  const float* of_b       = (const float*)d_in[23];
  const float* of_g       = (const float*)d_in[24];
  const float* of_be      = (const float*)d_in[25];
  float* out = (float*)d_out;

  const size_t per = (size_t)BB * TT * DD;    // elements per [B,T,D] buffer
  float* buf0 = (float*)d_ws;                 // impact -> state -> ctx
  float* lobE = buf0 + per;
  float* trdE = lobE + per;
  float* qbuf = trdE + per;                   // q -> fused
  float* kbuf = qbuf + per;
  float* vbuf = kbuf + per;

  const dim3 blk(256);
  const dim3 grd((BB * TT) / 128);

  // 1. impact = tanh(LN(trade @ ei_w^T + ei_b)) * has_trade
  gemm_ln_kernel<<<grd, blk, 0, stream>>>(trade_feat, nullptr, DD, 0,
      ei_w, ei_b, ei_g, ei_be, nullptr, has_trade, buf0, MODE_LN_TANH_MASK);
  // 2. state = EMA scan (in place)
  scan_kernel<<<dim3(4), blk, 0, stream>>>(buf0, decay);
  // 3. lob_enh = gelu(LN(concat(lob, state) @ lob_w^T + b))
  gemm_ln_kernel<<<grd, blk, 0, stream>>>(lob_feat, buf0, DD, DD,
      lob_w, lob_b, lob_g, lob_be, nullptr, nullptr, lobE, MODE_LN_GELU);
  // 4. trade_enh
  gemm_ln_kernel<<<grd, blk, 0, stream>>>(trade_feat, buf0, DD, DD,
      tr_w, tr_b, tr_g, tr_be, nullptr, nullptr, trdE, MODE_LN_GELU);
  // 5-7. q / k / v projections
  gemm_ln_kernel<<<grd, blk, 0, stream>>>(lobE, nullptr, DD, 0,
      in_proj_w, in_proj_b, nullptr, nullptr, nullptr, nullptr, qbuf, MODE_NONE);
  gemm_ln_kernel<<<grd, blk, 0, stream>>>(trdE, nullptr, DD, 0,
      in_proj_w + DD * DD, in_proj_b + DD,
      nullptr, nullptr, nullptr, nullptr, kbuf, MODE_NONE);
  gemm_ln_kernel<<<grd, blk, 0, stream>>>(trdE, nullptr, DD, 0,
      in_proj_w + 2 * DD * DD, in_proj_b + 2 * DD,
      nullptr, nullptr, nullptr, nullptr, vbuf, MODE_NONE);
  // 8. attention -> ctx (reuse buf0)
  attn_kernel<<<dim3(BB * NHH, TT / 128), blk, 0, stream>>>(qbuf, kbuf, vbuf, buf0);
  // 9. fused = LN(lob_enh + ctx @ out_proj^T + b)  (reuse qbuf)
  gemm_ln_kernel<<<grd, blk, 0, stream>>>(buf0, nullptr, DD, 0,
      out_proj_w, out_proj_b, cn_g, cn_b, lobE, nullptr, qbuf, MODE_RESID_LN);
  // 10. output = gelu(LN(concat(fused, trade_enh) @ of_w^T + b))
  gemm_ln_kernel<<<grd, blk, 0, stream>>>(qbuf, trdE, DD, DD,
      of_w, of_b, of_g, of_be, nullptr, nullptr, out, MODE_LN_GELU);

  (void)in_sizes; (void)n_in; (void)out_size; (void)ws_size;
}
